// DeepGCN_64579128263348
// MI455X (gfx1250) — matrix operand order repaired and checked by reading, not compiled, not run on hardware
//
#include <hip/hip_runtime.h>
#include <hip/hip_bf16.h>

// ---------------- problem constants ----------------
constexpr int NNODES  = 40000;
constexpr int NEDGES  = 400000;
constexpr int CIN     = 16;
constexpr int CDIM    = 64;
constexpr int CHID    = 128;
constexpr int NSTACK  = 7;            // L-1 stacked layers
constexpr int NC      = NNODES * CDIM;    // 2,560,000
constexpr int NCH     = NNODES * CHID;    // 5,120,000
constexpr int ECW     = NEDGES * CDIM;    // 25,600,000

#define EPS_MSG     1e-7f
#define BN_EPS_C    1e-5f
#define LRELU_SLOPE 0.01f

typedef _Float16 half_t;
typedef __attribute__((ext_vector_type(16))) _Float16 v16h;
typedef __attribute__((ext_vector_type(8)))  float    v8f;

// ---------------- helpers ----------------
__device__ __forceinline__ unsigned f2ord(float f) {
    unsigned u = __float_as_uint(f);
    return (u & 0x80000000u) ? ~u : (u | 0x80000000u);
}
__device__ __forceinline__ float ord2f(unsigned o) {
    unsigned u = (o & 0x80000000u) ? (o & 0x7fffffffu) : ~o;
    return __uint_as_float(u);
}

// ---------------- tiny utility kernels ----------------
__global__ void fill_f32_kernel(float* __restrict__ p, float v, int n) {
    int i = blockIdx.x * blockDim.x + threadIdx.x;
    if (i < n) p[i] = v;
}
__global__ void fill_u32_kernel(unsigned* __restrict__ p, unsigned v, int n) {
    int i = blockIdx.x * blockDim.x + threadIdx.x;
    if (i < n) p[i] = v;
}
__global__ void add_kernel(float* __restrict__ a, const float* __restrict__ b, int n) {
    int i = blockIdx.x * blockDim.x + threadIdx.x;
    if (i < n) a[i] += b[i];
}

// ---------------- edge phase: scatter softmax aggregation ----------------
__global__ void edge_max_kernel(const float* __restrict__ x, const float* __restrict__ ea,
                                const int* __restrict__ src, const int* __restrict__ dst,
                                const float* __restrict__ ew, const float* __restrict__ eb,
                                const float* __restrict__ tptr, unsigned* __restrict__ smax) {
    int i = blockIdx.x * blockDim.x + threadIdx.x;
    if (i >= ECW) return;
    int e = i >> 6, c = i & 63;
    float ef = ea[e] * ew[c] + eb[c];
    float m  = fmaxf(x[(size_t)src[e] * CDIM + c] + ef, 0.0f) + EPS_MSG;
    float s  = m * tptr[0];
    atomicMax(&smax[(size_t)dst[e] * CDIM + c], f2ord(s));
}

__global__ void edge_sum_kernel(const float* __restrict__ x, const float* __restrict__ ea,
                                const int* __restrict__ src, const int* __restrict__ dst,
                                const float* __restrict__ ew, const float* __restrict__ eb,
                                const float* __restrict__ tptr,
                                const unsigned* __restrict__ smax,
                                float* __restrict__ num, float* __restrict__ den) {
    int i = blockIdx.x * blockDim.x + threadIdx.x;
    if (i >= ECW) return;
    int e = i >> 6, c = i & 63;
    float ef = ea[e] * ew[c] + eb[c];
    float m  = fmaxf(x[(size_t)src[e] * CDIM + c] + ef, 0.0f) + EPS_MSG;
    float s  = m * tptr[0];
    size_t di = (size_t)dst[e] * CDIM + c;
    float ex = __expf(s - ord2f(smax[di]));
    atomicAdd(&den[di], ex);
    atomicAdd(&num[di], m * ex);
}

__global__ void finish_agg_kernel(float* __restrict__ xio, const float* __restrict__ num,
                                  const float* __restrict__ den, int n) {
    int i = blockIdx.x * blockDim.x + threadIdx.x;
    if (i >= n) return;
    float d = den[i];
    xio[i] += (d > 0.0f) ? (num[i] / d) : 0.0f;
}

// ---------------- BatchNorm (training mode, biased variance) ----------------
__global__ __launch_bounds__(256)
void bn_stats_kernel(const float* __restrict__ X, int nch, float* __restrict__ mv) {
    __shared__ float ssum[256];
    __shared__ float ssq[256];
    int ch = blockIdx.x;
    float s = 0.f, q = 0.f;
    for (int i = threadIdx.x; i < NNODES; i += 256) {
        float v = X[(size_t)i * nch + ch];
        s += v; q += v * v;
    }
    ssum[threadIdx.x] = s; ssq[threadIdx.x] = q;
    __syncthreads();
    for (int off = 128; off > 0; off >>= 1) {
        if (threadIdx.x < off) {
            ssum[threadIdx.x] += ssum[threadIdx.x + off];
            ssq[threadIdx.x]  += ssq[threadIdx.x + off];
        }
        __syncthreads();
    }
    if (threadIdx.x == 0) {
        float mean = ssum[0] / (float)NNODES;
        float var  = ssq[0] / (float)NNODES - mean * mean;
        mv[ch] = mean; mv[nch + ch] = var;
    }
}

__global__ void bn_apply_kernel(const float* __restrict__ in, float* __restrict__ out,
                                const float* __restrict__ mv, const float* __restrict__ g,
                                const float* __restrict__ b, int nch, int total, float slope) {
    int i = blockIdx.x * blockDim.x + threadIdx.x;
    if (i >= total) return;
    int c = i % nch;
    float y = (in[i] - mv[c]) * rsqrtf(mv[nch + c] + BN_EPS_C) * g[c] + b[c];
    out[i] = (y > 0.0f) ? y : y * slope;
}

// ---------------- WMMA GEMM: Out[M,Nc] = A[M,K] @ W[K,Nc] + bias ----------------
// K, Nc compile-time -> no divergent guards, vectorized fragment loads.
// One 16x16 output tile per wave32, K in 32-chunks via v_wmma_f32_16x16x32_f16.
//
// Fragment layouts per CDNA5 ISA 7.12.2 (wave32):
//   A 16x32 f16 : m = lane&15; lanes 0-15 k-base 0, lanes 16-31 k-base 8;
//                 frag[0..7] = A[m][k0+base+0..7], frag[8..15] = A[m][k0+16+base+0..7]
//   B 32x16 f16 : n = lane&15; same k order as A (lanes 0-15 k-base 0, lanes 16-31 k-base 8)
//                 -> W staged in LDS fragment-major: [k/32][(k/8)%2][n][(k/16)%2][k%8] so each lane's
//                    B fragment is one contiguous 32B LDS read (ds_load_b128 x2)
//   C/D 16x16 f32: vgpr j -> row j (lanes 0-15) / row j+8 (lanes 16-31), col = lane&15
template <int K, int Nc>
__global__ __launch_bounds__(256)
void gemm_bias_wmma(const float* __restrict__ A, const float* __restrict__ W,
                    const float* __restrict__ bias, float* __restrict__ Out, int M) {
    constexpr int KP = (K + 31) & ~31;                 // K padded to 32
    __shared__ half_t sW[KP * Nc];                     // fragment-major staged weights

    // stage + transpose weights: sW[(((k>>5)*2 + ((k>>3)&1))*Nc + n)*16 + (k&7) + ((k>>4)&1)*8] = W[k][n]
    for (int idx = threadIdx.x; idx < KP * Nc; idx += 256) {
        int k = idx / Nc, n = idx % Nc;
        float v = (k < K) ? W[k * Nc + n] : 0.0f;
        sW[((((k >> 5) * 2 + ((k >> 3) & 1)) * Nc) + n) * 16 + (k & 7) + ((k >> 4) & 1) * 8] = (half_t)v;
    }
    __syncthreads();

    constexpr int tilesN = Nc >> 4;
    const int totalTiles = (M >> 4) * tilesN;
    const int tileId     = blockIdx.x * 8 + (threadIdx.x >> 5);
    if (tileId >= totalTiles) return;

    const int mBase = (tileId / tilesN) << 4;
    const int nBase = (tileId % tilesN) << 4;

    const int lane = threadIdx.x & 31;
    const int hsel = lane >> 4;      // lane-half select
    const int lidx = lane & 15;
    const int col  = nBase + lidx;
    const int row  = mBase + lidx;
    const float* __restrict__ Arow = A + (size_t)row * K;

    v8f acc;
    const float bv = bias[col];
#pragma unroll
    for (int j = 0; j < 8; ++j) acc[j] = bv;

#pragma unroll
    for (int k0 = 0; k0 < KP; k0 += 32) {
        if (k0 + 32 < K) __builtin_prefetch(Arow + k0 + 32, 0, 3);  // global_prefetch_b8

        // ---- A fragment: two 16-float half-chunks, each 2x float4 ----
        v16h afrag;
        {
            const float4 p0 = *(const float4*)(Arow + k0 + hsel * 8);
            const float4 p1 = *(const float4*)(Arow + k0 + hsel * 8 + 4);
            afrag[0] = (half_t)p0.x; afrag[1] = (half_t)p0.y;
            afrag[2] = (half_t)p0.z; afrag[3] = (half_t)p0.w;
            afrag[4] = (half_t)p1.x; afrag[5] = (half_t)p1.y;
            afrag[6] = (half_t)p1.z; afrag[7] = (half_t)p1.w;
        }
        if (k0 + 16 < K) {           // uniform, constant-folds after unroll
            const float4 q0 = *(const float4*)(Arow + k0 + 16 + hsel * 8);
            const float4 q1 = *(const float4*)(Arow + k0 + 16 + hsel * 8 + 4);
            afrag[8]  = (half_t)q0.x; afrag[9]  = (half_t)q0.y;
            afrag[10] = (half_t)q0.z; afrag[11] = (half_t)q0.w;
            afrag[12] = (half_t)q1.x; afrag[13] = (half_t)q1.y;
            afrag[14] = (half_t)q1.z; afrag[15] = (half_t)q1.w;
        } else {
#pragma unroll
            for (int j = 8; j < 16; ++j) afrag[j] = (half_t)0.0f;
        }

        // ---- B fragment: one contiguous 32-byte LDS read ----
        const v16h bfrag =
            *(const v16h*)&sW[((((k0 >> 4) + hsel) * Nc) + col) * 16];

        acc = __builtin_amdgcn_wmma_f32_16x16x32_f16(
            /*neg_a=*/false, afrag, /*neg_b=*/false, bfrag,
            /*c_mod=*/(short)0, acc, /*reuse_a=*/false, /*reuse_b=*/false);
    }

#pragma unroll
    for (int j = 0; j < 8; ++j) {
        int r = mBase + hsel * 8 + j;
        Out[(size_t)r * Nc + col] = acc[j];
    }
}

// ---------------- host orchestration ----------------
extern "C" void kernel_launch(void* const* d_in, const int* in_sizes, int n_in,
                              void* d_out, int out_size, void* d_ws, size_t ws_size,
                              hipStream_t stream) {
    (void)in_sizes; (void)n_in; (void)out_size; (void)ws_size;
    const float* x       = (const float*)d_in[0];
    const float* ea      = (const float*)d_in[1];
    const int*   ei      = (const int*)  d_in[2];
    // d_in[3] = batch (unused: single graph)
    const float* lin0_w  = (const float*)d_in[4];
    const float* lin0_b  = (const float*)d_in[5];
    const float* e0_w    = (const float*)d_in[6];
    const float* e0_b    = (const float*)d_in[7];
    const float* t0      = (const float*)d_in[8];
    const float* m0_w1   = (const float*)d_in[9];
    const float* m0_b1   = (const float*)d_in[10];
    const float* m0_g1   = (const float*)d_in[11];
    const float* m0_be1  = (const float*)d_in[12];
    const float* m0_w2   = (const float*)d_in[13];
    const float* m0_b2   = (const float*)d_in[14];
    const float* norm0_g = (const float*)d_in[15];
    const float* norm0_b = (const float*)d_in[16];
    const float* le_w    = (const float*)d_in[17];
    const float* le_b    = (const float*)d_in[18];
    const float* lt      = (const float*)d_in[19];
    const float* lw1     = (const float*)d_in[20];
    const float* lb1     = (const float*)d_in[21];
    const float* lg1     = (const float*)d_in[22];
    const float* lbe1    = (const float*)d_in[23];
    const float* lw2     = (const float*)d_in[24];
    const float* lb2     = (const float*)d_in[25];
    const float* lng     = (const float*)d_in[26];
    const float* lnb     = (const float*)d_in[27];

    const int* srcp = ei;
    const int* dstp = ei + NEDGES;

    // workspace layout (floats): 6*NC + NCH + 256  ~= 82 MB
    float* ws      = (float*)d_ws;
    float* h       = ws;                         // [N,C] running residual state
    float* xin     = ws + (size_t)NC;            // [N,C] genconv input / agg out
    float* fnum    = ws + (size_t)2 * NC;        // [N,C] softmax numerator
    float* fden    = ws + (size_t)3 * NC;        // [N,C] softmax denominator
    unsigned* smax = (unsigned*)(ws + (size_t)4 * NC);  // [N,C] ordered-float max
    float* t1      = ws + (size_t)5 * NC;        // [N,CHID] MLP hidden
    float* stats   = t1 + (size_t)NCH;           // 2*CHID mean|var
    float* gout    = stats + 2 * CHID;           // [N,C] genconv output

    const int ncb = (NC + 255) / 256;
    const int ecb = (ECW + 255) / 256;

    // GEMM grids: one 16x16 tile per wave, 8 waves/block
    const int blk_lin0 = ((NNODES / 16) * (CDIM / 16) + 7) / 8;   // K=16 ,Nc=64
    const int blk_mlp1 = ((NNODES / 16) * (CHID / 16) + 7) / 8;   // K=64 ,Nc=128
    const int blk_mlp2 = ((NNODES / 16) * (CDIM / 16) + 7) / 8;   // K=128,Nc=64

    auto genconv = [&](const float* ew, const float* ebv, const float* tv,
                       const float* w1, const float* b1, const float* g1, const float* be1,
                       const float* w2, const float* b2, float* xio, float* outb) {
        fill_u32_kernel<<<ncb, 256, 0, stream>>>(smax, 0u, NC);
        fill_f32_kernel<<<ncb, 256, 0, stream>>>(fnum, 0.0f, NC);
        fill_f32_kernel<<<ncb, 256, 0, stream>>>(fden, 0.0f, NC);
        edge_max_kernel<<<ecb, 256, 0, stream>>>(xio, ea, srcp, dstp, ew, ebv, tv, smax);
        edge_sum_kernel<<<ecb, 256, 0, stream>>>(xio, ea, srcp, dstp, ew, ebv, tv, smax,
                                                 fnum, fden);
        finish_agg_kernel<<<ncb, 256, 0, stream>>>(xio, fnum, fden, NC);
        // MLP: C -> 2C (bn+relu) -> C on WMMA
        gemm_bias_wmma<CDIM, CHID><<<blk_mlp1, 256, 0, stream>>>(xio, w1, b1, t1, NNODES);
        bn_stats_kernel<<<CHID, 256, 0, stream>>>(t1, CHID, stats);
        bn_apply_kernel<<<(NCH + 255) / 256, 256, 0, stream>>>(t1, t1, stats, g1, be1,
                                                               CHID, NCH, 0.0f /*relu*/);
        gemm_bias_wmma<CHID, CDIM><<<blk_mlp2, 256, 0, stream>>>(t1, w2, b2, outb, NNODES);
    };

    // layer 0: lin_src projection then GENConv -> h
    gemm_bias_wmma<CIN, CDIM><<<blk_lin0, 256, 0, stream>>>(x, lin0_w, lin0_b, xin, NNODES);
    genconv(e0_w, e0_b, t0, m0_w1, m0_b1, m0_g1, m0_be1, m0_w2, m0_b2, xin, h);

    // res+ blocks 1..L-1: z = leaky(bn(h)); z = genconv(z); h += z
    for (int i = 0; i < NSTACK; ++i) {
        bn_stats_kernel<<<CDIM, 256, 0, stream>>>(h, CDIM, stats);
        bn_apply_kernel<<<ncb, 256, 0, stream>>>(h, xin, stats, lng + i * CDIM,
                                                 lnb + i * CDIM, CDIM, NC, LRELU_SLOPE);
        genconv(le_w + i * CDIM, le_b + i * CDIM, lt + i,
                lw1 + (size_t)i * CDIM * CHID, lb1 + i * CHID,
                lg1 + i * CHID, lbe1 + i * CHID,
                lw2 + (size_t)i * CHID * CDIM, lb2 + i * CDIM, xin, gout);
        add_kernel<<<ncb, 256, 0, stream>>>(h, gout, NC);
    }

    // final: leaky(bn(h, norm0)) -> d_out
    bn_stats_kernel<<<CDIM, 256, 0, stream>>>(h, CDIM, stats);
    bn_apply_kernel<<<ncb, 256, 0, stream>>>(h, (float*)d_out, stats, norm0_g, norm0_b,
                                             CDIM, NC, LRELU_SLOPE);
}
